// SteelPlateAttentionModel_42365557407998
// MI455X (gfx1250) — matrix operand order repaired and checked by reading, not compiled, not run on hardware
//
#include <hip/hip_runtime.h>
#include <hip/hip_bf16.h>

#define NN     50000
#define NE     800000
#define FIN    45
#define EMB    128
#define NHEADS 4
#define HDIM   32
#define NB     64
#define NACT   25

typedef __attribute__((ext_vector_type(16))) __bf16 v16bf;
typedef __attribute__((ext_vector_type(8)))  float  v8f;

static inline int cdiv(int a, int b) { return (a + b - 1) / b; }

// ---------------------------------------------------------------- device utils
__device__ __forceinline__ __bf16 f2bf(float f) { return (__bf16)f; }

__device__ __forceinline__ void atomicMaxF(float* addr, float v) {
    unsigned int* a = (unsigned int*)addr;
    unsigned int old = *a;
    while (true) {
        float f = __uint_as_float(old);
        if (f >= v) break;
        unsigned int prev = atomicCAS(a, old, __float_as_uint(v));
        if (prev == old) break;
        old = prev;
    }
}

__device__ __forceinline__ float waveSum(float v) {
    #pragma unroll
    for (int m = 16; m; m >>= 1) v += __shfl_xor(v, m, 32);
    return v;
}
__device__ __forceinline__ float waveMax(float v) {
    #pragma unroll
    for (int m = 16; m; m >>= 1) v = fmaxf(v, __shfl_xor(v, m, 32));
    return v;
}

// ---------------------------------------------------------------- WMMA GEMM
// C[h] = A[h] (MxK row-major, ld lda) @ W[h] (KxN, ld ldw) + bias.
// One wave32 per 16x16 tile. K is a template constant -> K-loop fully unrolls.
// OOB rows/cols are CLAMPED (not predicated): garbage lands in accumulator
// lanes that are never stored, so the hot loop has no EXEC manipulation.
template <int K>
__global__ void wmma_gemm_bias(const float* __restrict__ A, int lda, int aH,
                               const float* __restrict__ W, int ldw, int wH,
                               const float* __restrict__ bias,
                               float* __restrict__ C, int ldc, int cH,
                               int M, int N) {
    const int lane   = threadIdx.x;
    const int tilesN = (N + 15) >> 4;
    const int tilesM = (M + 15) >> 4;
    const int tile   = blockIdx.x * blockDim.y + threadIdx.y;
    if (tile >= tilesM * tilesN) return;            // wave-uniform exit
    const int tm = tile / tilesN, tn = tile % tilesN;

    const float* Ah = A + (long)blockIdx.z * aH;
    const float* Wh = W + (long)blockIdx.z * wH;
    float*       Ch = C + (long)blockIdx.z * cH;

    const int half = lane >> 4;
    const int l16  = lane & 15;
    const int rowA = tm * 16 + l16;
    const int colB = tn * 16 + l16;
    const int rowC = (rowA < M) ? rowA : (M - 1);   // clamped (memory-safe)
    const int colC = (colB < N) ? colB : (N - 1);

    const float* arow = Ah + (long)rowC * lda;
    const bool vecOK = ((lda & 3) == 0);            // wave-uniform

    v8f acc = {};
    #pragma unroll
    for (int k0 = 0; k0 < K; k0 += 32) {
        v16bf a, b;
        if ((k0 + 32 <= K) && vecOK) {
            // full K-step, 16B-aligned rows: A fragment = 4 x b128 loads
            const float4* p0 = (const float4*)(arow + k0 + (half << 3));
            const float4* p1 = (const float4*)(arow + k0 + 16 + (half << 3));
            float4 a0 = p0[0], a1 = p0[1], a2 = p1[0], a3 = p1[1];
            a[0]  = f2bf(a0.x); a[1]  = f2bf(a0.y); a[2]  = f2bf(a0.z); a[3]  = f2bf(a0.w);
            a[4]  = f2bf(a1.x); a[5]  = f2bf(a1.y); a[6]  = f2bf(a1.z); a[7]  = f2bf(a1.w);
            a[8]  = f2bf(a2.x); a[9]  = f2bf(a2.y); a[10] = f2bf(a2.z); a[11] = f2bf(a2.w);
            a[12] = f2bf(a3.x); a[13] = f2bf(a3.y); a[14] = f2bf(a3.z); a[15] = f2bf(a3.w);
        } else {
            #pragma unroll
            for (int i = 0; i < 16; ++i) {
                int ka = k0 + ((i >> 3) << 4) + (half << 3) + (i & 7);
                float av = arow[ka < K ? ka : (K - 1)];   // unconditional load
                a[i] = f2bf((ka < K) ? av : 0.0f);        // value select only
            }
        }
        #pragma unroll
        for (int i = 0; i < 16; ++i) {
            int kb = k0 + ((i >> 3) << 4) + (half << 3) + (i & 7);
            float bv = Wh[(long)(kb < K ? kb : (K - 1)) * ldw + colC];
            b[i] = f2bf((kb < K) ? bv : 0.0f);
        }
        acc = __builtin_amdgcn_wmma_f32_16x16x32_bf16(false, a, false, b,
                                                      (short)0, acc, false, false);
    }

    const float bc = bias ? bias[colC] : 0.0f;
    float* cp = Ch + (long)(tm * 16 + half * 8) * ldc + colB;
    if (tm * 16 + 16 <= M) {                        // wave-uniform full-M path
        if (colB < N) {                             // single EXEC region, 8 stores
            #pragma unroll
            for (int r = 0; r < 8; ++r)
                cp[(long)r * ldc] = acc[r] + bc;
        }
    } else {
        #pragma unroll
        for (int r = 0; r < 8; ++r) {
            int row = tm * 16 + r + half * 8;
            if (row < M && colB < N)
                cp[(long)r * ldc] = acc[r] + bc;
        }
    }
}

// ---------------------------------------------------------------- elementwise
__global__ void fill_kernel(float* p, float v, int n) {
    int i = blockIdx.x * blockDim.x + threadIdx.x;
    if (i < n) p[i] = v;
}
__global__ void add_kernel(float* a, const float* b, int n) {
    int i = blockIdx.x * blockDim.x + threadIdx.x;
    if (i < n) a[i] += b[i];
}
__global__ void gelu_kernel(float* x, int n) {
    int i = blockIdx.x * blockDim.x + threadIdx.x;
    if (i < n) { float v = x[i]; x[i] = 0.5f * v * (1.0f + erff(v * 0.70710678118654752f)); }
}
// layer-2 fusion of HGT skip gate + outer residual: out = s*out + (1-s)*x + x
__global__ void skipres_kernel(float* out, const float* x, const float* skipp, int n) {
    int i = blockIdx.x * blockDim.x + threadIdx.x;
    if (i < n) {
        float s = 1.0f / (1.0f + expf(-skipp[0]));
        out[i] = s * out[i] + (2.0f - s) * x[i];
    }
}

// LayerNorm (+optional relu), one wave per row
__global__ void ln_act_kernel(const float* __restrict__ in, const float* __restrict__ g,
                              const float* __restrict__ b, float* __restrict__ out,
                              int rows, int D, int relu) {
    int row = blockIdx.x * blockDim.y + threadIdx.y;
    if (row >= rows) return;
    int lane = threadIdx.x;
    const float* r = in + (long)row * D;
    float s = 0.f, s2 = 0.f;
    for (int j = lane; j < D; j += 32) { float v = r[j]; s += v; s2 += v * v; }
    s = waveSum(s); s2 = waveSum(s2);
    float mean = s / D;
    float var  = s2 / D - mean * mean;
    float inv  = rsqrtf(var + 1e-5f);
    float* o = out + (long)row * D;
    for (int j = lane; j < D; j += 32) {
        float v = (r[j] - mean) * inv * g[j] + b[j];
        if (relu) v = fmaxf(v, 0.f);
        o[j] = v;
    }
}

// ---------------------------------------------------------------- edge phase
__global__ void edge_alpha_kernel(const int* __restrict__ ei, const float* __restrict__ q,
                                  const float* __restrict__ k2, const float* __restrict__ p_rel,
                                  float* __restrict__ alpha, float* __restrict__ amax, int E) {
    int t = blockIdx.x * blockDim.x + threadIdx.x;
    if (t >= E * NHEADS) return;
    int e = t >> 2, h = t & 3;
    int src = ei[e], dst = ei[E + e];
    const float* qp = q  + (long)dst * EMB + h * HDIM;
    const float* kp = k2 + (long)src * EMB + h * HDIM;
    float acc = 0.f;
    #pragma unroll
    for (int i = 0; i < HDIM; i += 4) {
        float4 qa = *(const float4*)(qp + i);
        float4 kb = *(const float4*)(kp + i);
        acc += qa.x * kb.x + qa.y * kb.y + qa.z * kb.z + qa.w * kb.w;
    }
    float al = acc * p_rel[h] * 0.17677669529663687f;   // 1/sqrt(32)
    alpha[t] = al;
    atomicMaxF(&amax[dst * NHEADS + h], al);
}

__global__ void edge_exp_kernel(const int* __restrict__ ei, float* __restrict__ alpha,
                                const float* __restrict__ amax, float* __restrict__ asum, int E) {
    int t = blockIdx.x * blockDim.x + threadIdx.x;
    if (t >= E * NHEADS) return;
    int e = t >> 2, h = t & 3;
    int dst = ei[E + e];
    float x = expf(alpha[t] - amax[dst * NHEADS + h]);
    alpha[t] = x;                                        // in-place: alpha -> aexp
    atomicAdd(&asum[dst * NHEADS + h], x);
}

__global__ void edge_scatter_kernel(const int* __restrict__ ei, const float* __restrict__ aexp,
                                    const float* __restrict__ asum, const float* __restrict__ v2,
                                    float* __restrict__ agg, int E) {
    long t = (long)blockIdx.x * blockDim.x + threadIdx.x;
    if (t >= (long)E * EMB) return;
    int d = (int)(t & (EMB - 1));
    int e = (int)(t >> 7);
    int h = d >> 5;
    int src = ei[e], dst = ei[E + e];
    float a = aexp[e * NHEADS + h] / (asum[dst * NHEADS + h] + 1e-16f);
    atomicAdd(&agg[(long)dst * EMB + d], v2[(long)src * EMB + d] * a);
}

// ---------------------------------------------------------------- pooling
__global__ void rowdot_kernel(const float* __restrict__ H, const float* __restrict__ w,
                              const float* __restrict__ bias, float* __restrict__ out,
                              int rows, int D) {
    int n = blockIdx.x * blockDim.y + threadIdx.y;
    if (n >= rows) return;
    int lane = threadIdx.x;
    float s = 0.f;
    for (int j = lane; j < D; j += 32) s += H[(long)n * D + j] * w[j];
    s = waveSum(s);
    if (lane == 0) out[n] = s + bias[0];
}

__global__ void pool1_max_kernel(const float* __restrict__ gate, const float* __restrict__ X,
                                 const float* __restrict__ wob, const float* __restrict__ bob,
                                 const int* __restrict__ batch, float* __restrict__ bmax, int Nn) {
    int n = blockIdx.x * blockDim.y + threadIdx.y;
    if (n >= Nn) return;
    int lane = threadIdx.x;
    float gn = gate[n], x0 = X[(long)n * FIN];
    float mx = -3.402823466e38f;
    for (int j = lane; j < EMB; j += 32) mx = fmaxf(mx, gn + x0 * wob[j] + bob[j]);
    mx = waveMax(mx);
    if (lane == 0) atomicMaxF(&bmax[batch[n]], mx);
}

__global__ void pool1_sum_kernel(const float* __restrict__ gate, const float* __restrict__ X,
                                 const float* __restrict__ wob, const float* __restrict__ bob,
                                 const int* __restrict__ batch, const float* __restrict__ bmax,
                                 float* __restrict__ bsum, int Nn) {
    int n = blockIdx.x * blockDim.y + threadIdx.y;
    if (n >= Nn) return;
    int lane = threadIdx.x;
    int b = batch[n];
    float gn = gate[n], x0 = X[(long)n * FIN], m = bmax[b];
    float s = 0.f;
    for (int j = lane; j < EMB; j += 32) s += expf(gn + x0 * wob[j] + bob[j] - m);
    s = waveSum(s);
    if (lane == 0) atomicAdd(&bsum[b], s);
}

__global__ void pool1_scatter_kernel(const float* __restrict__ gate, const float* __restrict__ X,
                                     const float* __restrict__ wob, const float* __restrict__ bob,
                                     const int* __restrict__ batch, const float* __restrict__ bmax,
                                     const float* __restrict__ bsum, const float* __restrict__ H,
                                     float* __restrict__ pooled, int Nn) {
    long t = (long)blockIdx.x * blockDim.x + threadIdx.x;
    if (t >= (long)Nn * EMB) return;
    int j = (int)(t & (EMB - 1));
    int n = (int)(t >> 7);
    int b = batch[n];
    float e = expf(gate[n] + X[(long)n * FIN] * wob[j] + bob[j] - bmax[b]);
    float attn = e / (bsum[b] + 1e-12f);
    atomicAdd(&pooled[b * EMB + j], attn * H[(long)n * EMB + j]);
}

__global__ void score_kernel(const float* __restrict__ kg, const float* __restrict__ X,
                             const float* __restrict__ wob, const float* __restrict__ bob,
                             const float* __restrict__ qv, const int* __restrict__ batch,
                             float* __restrict__ scores, float* __restrict__ bmax2, int Nn) {
    int n = blockIdx.x * blockDim.y + threadIdx.y;
    if (n >= Nn) return;
    int lane = threadIdx.x;
    int b = batch[n];
    float x0 = X[(long)n * FIN];
    float s = 0.f;
    for (int j = lane; j < EMB; j += 32) {
        float ob = x0 * wob[j] + bob[j];
        s += (kg[(long)n * EMB + j] + ob) * qv[b * EMB + j];
    }
    s = waveSum(s);
    if (lane == 0) {
        float sc = s * 0.08838834764831845f;             // 1/sqrt(128)
        scores[n] = sc;
        atomicMaxF(&bmax2[b], sc);
    }
}

__global__ void score_exp_kernel(const float* __restrict__ scores, const int* __restrict__ batch,
                                 const float* __restrict__ bmax2, float* __restrict__ se,
                                 float* __restrict__ bsum2, int Nn) {
    int n = blockIdx.x * blockDim.x + threadIdx.x;
    if (n >= Nn) return;
    int b = batch[n];
    float v = expf(scores[n] - bmax2[b]);
    se[n] = v;
    atomicAdd(&bsum2[b], v);
}

__global__ void attn_scatter_kernel(const float* __restrict__ se, const float* __restrict__ bsum2,
                                    const int* __restrict__ batch, const float* __restrict__ H,
                                    float* __restrict__ pooled2, float* __restrict__ attn_out, int Nn) {
    long t = (long)blockIdx.x * blockDim.x + threadIdx.x;
    if (t >= (long)Nn * EMB) return;
    int j = (int)(t & (EMB - 1));
    int n = (int)(t >> 7);
    int b = batch[n];
    float a = se[n] / bsum2[b];
    atomicAdd(&pooled2[b * EMB + j], H[(long)n * EMB + j] * a);
    if (j == 0) attn_out[n] = a;
}

__global__ void softmax_rows_kernel(const float* __restrict__ in, float* __restrict__ out,
                                    int rows, int D) {
    int row = blockIdx.x * blockDim.y + threadIdx.y;
    if (row >= rows) return;
    int lane = threadIdx.x;
    const float* r = in + (long)row * D;
    float mx = -3.402823466e38f;
    for (int j = lane; j < D; j += 32) mx = fmaxf(mx, r[j]);
    mx = waveMax(mx);
    float s = 0.f;
    for (int j = lane; j < D; j += 32) s += expf(r[j] - mx);
    s = waveSum(s);
    for (int j = lane; j < D; j += 32) out[(long)row * D + j] = expf(r[j] - mx) / s;
}

// ---------------------------------------------------------------- host helpers
template <int K>
static void gemm_t(hipStream_t s, const float* A, int lda, int aH,
                   const float* W, int ldw, int wH, const float* bias,
                   float* C, int ldc, int cH, int M, int N, int heads) {
    int tiles = cdiv(M, 16) * cdiv(N, 16);
    dim3 blk(32, 4, 1), grd(cdiv(tiles, 4), 1, heads);
    wmma_gemm_bias<K><<<grd, blk, 0, s>>>(A, lda, aH, W, ldw, wH, bias, C, ldc, cH, M, N);
}
static void gemm(hipStream_t s, const float* A, int lda, int aH,
                 const float* W, int ldw, int wH, const float* bias,
                 float* C, int ldc, int cH, int M, int K, int N, int heads) {
    if      (K == FIN)     gemm_t<FIN>    (s, A, lda, aH, W, ldw, wH, bias, C, ldc, cH, M, N, heads);
    else if (K == HDIM)    gemm_t<HDIM>   (s, A, lda, aH, W, ldw, wH, bias, C, ldc, cH, M, N, heads);
    else if (K == EMB)     gemm_t<EMB>    (s, A, lda, aH, W, ldw, wH, bias, C, ldc, cH, M, N, heads);
    else if (K == 2 * EMB) gemm_t<2 * EMB>(s, A, lda, aH, W, ldw, wH, bias, C, ldc, cH, M, N, heads);
}
static void ln(hipStream_t s, const float* in, const float* g, const float* b,
               float* out, int rows, int D, int relu) {
    dim3 blk(32, 8), grd(cdiv(rows, 8));
    ln_act_kernel<<<grd, blk, 0, s>>>(in, g, b, out, rows, D, relu);
}
static void fill(hipStream_t s, float* p, float v, int n) {
    fill_kernel<<<cdiv(n, 256), 256, 0, s>>>(p, v, n);
}

struct HgtP { const float *a_rel,*ba,*bk,*bq,*bv,*m_rel,*p_rel,*skip,*wa,*wk,*wq,*wv; };

static void run_hgt_layer(hipStream_t s, const float* xin, int Fin, const HgtP& P,
                          const float* ln_g, const float* ln_b, float* hout,
                          float* kb, float* qb, float* vb, float* k2, float* v2,
                          float* agg, float* hn, float* alpha, float* amax, float* asum,
                          const int* EI, bool skipres) {
    gemm(s, xin, Fin, 0, P.wk, EMB, 0, P.bk, kb, EMB, 0, NN, Fin, EMB, 1);
    gemm(s, xin, Fin, 0, P.wq, EMB, 0, P.bq, qb, EMB, 0, NN, Fin, EMB, 1);
    gemm(s, xin, Fin, 0, P.wv, EMB, 0, P.bv, vb, EMB, 0, NN, Fin, EMB, 1);
    // per-head einsums: k2[n,h,:] = k[n,h,:] @ a_rel[h] ; v2 with m_rel
    gemm(s, kb, EMB, HDIM, P.a_rel, HDIM, HDIM * HDIM, nullptr, k2, EMB, HDIM, NN, HDIM, HDIM, NHEADS);
    gemm(s, vb, EMB, HDIM, P.m_rel, HDIM, HDIM * HDIM, nullptr, v2, EMB, HDIM, NN, HDIM, HDIM, NHEADS);
    fill(s, amax, -3.402823466e38f, NN * NHEADS);
    fill(s, asum, 0.f, NN * NHEADS);
    fill(s, agg, 0.f, NN * EMB);
    edge_alpha_kernel<<<cdiv(NE * NHEADS, 256), 256, 0, s>>>(EI, qb, k2, P.p_rel, alpha, amax, NE);
    edge_exp_kernel<<<cdiv(NE * NHEADS, 256), 256, 0, s>>>(EI, alpha, amax, asum, NE);
    long tot = (long)NE * EMB;
    edge_scatter_kernel<<<(int)((tot + 255) / 256), 256, 0, s>>>(EI, alpha, asum, v2, agg, NE);
    gelu_kernel<<<cdiv(NN * EMB, 256), 256, 0, s>>>(agg, NN * EMB);
    gemm(s, agg, EMB, 0, P.wa, EMB, 0, P.ba, hn, EMB, 0, NN, EMB, EMB, 1);
    if (skipres) skipres_kernel<<<cdiv(NN * EMB, 256), 256, 0, s>>>(hn, xin, P.skip, NN * EMB);
    ln(s, hn, ln_g, ln_b, hout, NN, EMB, 1);
}

struct Blk { const float *b,*be,*g,*rb,*rbe,*rg,*rw,*w; };

static const float* run_block(hipStream_t s, const float* xin, int Din, int Dout, const Blk& B,
                              float* t1, float* t2, int M) {
    gemm(s, xin, Din, 0, B.w, Dout, 0, B.b, t1, Dout, 0, M, Din, Dout, 1);
    ln(s, t1, B.g, B.be, t1, M, Dout, 1);
    if (B.rw) {
        gemm(s, t1, Dout, 0, B.rw, Dout, 0, B.rb, t2, Dout, 0, M, Dout, Dout, 1);
        ln(s, t2, B.rg, B.rbe, t2, M, Dout, 1);
        add_kernel<<<cdiv(M * Dout, 256), 256, 0, s>>>(t1, t2, M * Dout);
    }
    return t1;
}

// ---------------------------------------------------------------- entry point
extern "C" void kernel_launch(void* const* d_in, const int* in_sizes, int n_in,
                              void* d_out, int out_size, void* d_ws, size_t ws_size,
                              hipStream_t stream) {
    (void)in_sizes; (void)n_in; (void)out_size; (void)ws_size;
    hipStream_t s = stream;

    const float* X     = (const float*)d_in[0];
    const int*   EI    = (const int*)d_in[1];
    const int*   BATCH = (const int*)d_in[2];
    int ip = 3;
    auto nxt = [&]() { return (const float*)d_in[ip++]; };
    auto rdBlk = [&](bool res) {
        Blk b{};
        b.b = nxt(); b.be = nxt(); b.g = nxt();
        if (res) { b.rb = nxt(); b.rbe = nxt(); b.rg = nxt(); b.rw = nxt(); }
        b.w = nxt();
        return b;
    };
    // params pytree, sorted-key order: actor, c1, c2, gate, gk, gq, gv, hgt, lns, obp, proj
    Blk aB0 = rdBlk(false), aB1 = rdBlk(false), aB2 = rdBlk(true), aB3 = rdBlk(true);
    const float* actor_fb = nxt(); const float* actor_fw = nxt();
    Blk c1B0 = rdBlk(true), c1B1 = rdBlk(true), c1B2 = rdBlk(true);
    const float* c1_fb = nxt(); const float* c1_fw = nxt();
    Blk c2B0 = rdBlk(true), c2B1 = rdBlk(true), c2B2 = rdBlk(true);
    const float* c2_fb = nxt(); const float* c2_fw = nxt();
    const float* gate_b = nxt(); const float* gate_w = nxt();
    const float* gk_b = nxt();   const float* gk_w = nxt();
    const float* gq_b = nxt();   const float* gq_w = nxt();
    nxt(); nxt();                                  // gv (unused in forward)
    auto rdHgt = [&]() {
        HgtP h;
        h.a_rel = nxt(); h.ba = nxt(); h.bk = nxt(); h.bq = nxt(); h.bv = nxt();
        h.m_rel = nxt(); h.p_rel = nxt(); h.skip = nxt();
        h.wa = nxt(); h.wk = nxt(); h.wq = nxt(); h.wv = nxt();
        return h;
    };
    HgtP hgt0 = rdHgt(), hgt1 = rdHgt();
    const float* ln0_b = nxt(); const float* ln0_g = nxt();
    const float* ln1_b = nxt(); const float* ln1_g = nxt();
    const float* ob_b = nxt();  const float* ob_w = nxt();
    const float* pj_b = nxt();  const float* pj_w = nxt();

    // workspace carve-out
    char* wsp = (char*)d_ws; size_t off = 0;
    auto alloc = [&](size_t elems) {
        off = (off + 255) & ~size_t(255);
        float* p = (float*)(wsp + off);
        off += elems * sizeof(float);
        return p;
    };
    float* h    = alloc((size_t)NN * EMB);
    float* hn   = alloc((size_t)NN * EMB);
    float* kb   = alloc((size_t)NN * EMB);
    float* qb   = alloc((size_t)NN * EMB);
    float* vb   = alloc((size_t)NN * EMB);
    float* k2   = alloc((size_t)NN * EMB);
    float* v2   = alloc((size_t)NN * EMB);
    float* agg  = alloc((size_t)NN * EMB);
    float* alpha= alloc((size_t)NE * NHEADS);
    float* amax = alloc((size_t)NN * NHEADS);
    float* asum = alloc((size_t)NN * NHEADS);
    float* gateb= alloc(NN);
    float* scr  = alloc(NN);
    float* se   = alloc(NN);
    float* bmax = alloc(NB);  float* bsum = alloc(NB);
    float* bmax2= alloc(NB);  float* bsum2= alloc(NB);
    float* pooled  = alloc(NB * EMB);
    float* qv      = alloc(NB * EMB);
    float* pooled2 = alloc(NB * EMB);
    float* gc      = alloc(NB * EMB);
    float* gc2     = alloc(NB * EMB);
    float* t0 = alloc(NB * 2 * EMB);
    float* t1 = alloc(NB * 2 * EMB);
    float* t2 = alloc(NB * 2 * EMB);
    float* logits = alloc(NB * NACT);

    // ---- HGT layer 0 (no skip/residual: 45 != 128), layer 1 (skip + residual)
    run_hgt_layer(s, X, FIN, hgt0, ln0_g, ln0_b, h,
                  kb, qb, vb, k2, v2, agg, hn, alpha, amax, asum, EI, false);
    run_hgt_layer(s, h, EMB, hgt1, ln1_g, ln1_b, hn,
                  kb, qb, vb, k2, v2, agg, agg /*reuse as hn*/, alpha, amax, asum, EI, true);
    float* H = hn;

    // ---- gated pooling 1
    dim3 wblk(32, 8);
    rowdot_kernel<<<cdiv(NN, 8), wblk, 0, s>>>(H, gate_w, gate_b, gateb, NN, EMB);
    fill(s, bmax, -3.402823466e38f, NB);
    fill(s, bsum, 0.f, NB);
    fill(s, pooled, 0.f, NB * EMB);
    pool1_max_kernel<<<cdiv(NN, 8), wblk, 0, s>>>(gateb, X, ob_w, ob_b, BATCH, bmax, NN);
    pool1_sum_kernel<<<cdiv(NN, 8), wblk, 0, s>>>(gateb, X, ob_w, ob_b, BATCH, bmax, bsum, NN);
    pool1_scatter_kernel<<<cdiv(NN * EMB, 256), 256, 0, s>>>(gateb, X, ob_w, ob_b, BATCH,
                                                             bmax, bsum, H, pooled, NN);
    gemm(s, pooled, EMB, 0, pj_w, EMB, 0, pj_b, gc, EMB, 0, NB, EMB, EMB, 1);
    gemm(s, gc, EMB, 0, gq_w, EMB, 0, gq_b, qv, EMB, 0, NB, EMB, EMB, 1);
    gemm(s, H, EMB, 0, gk_w, EMB, 0, gk_b, kb /*reuse as kg*/, EMB, 0, NN, EMB, EMB, 1);

    // ---- scored pooling 2
    fill(s, bmax2, -3.402823466e38f, NB);
    fill(s, bsum2, 0.f, NB);
    fill(s, pooled2, 0.f, NB * EMB);
    score_kernel<<<cdiv(NN, 8), wblk, 0, s>>>(kb, X, ob_w, ob_b, qv, BATCH, scr, bmax2, NN);
    score_exp_kernel<<<cdiv(NN, 256), 256, 0, s>>>(scr, BATCH, bmax2, se, bsum2, NN);
    float* out_f = (float*)d_out;
    float* attn_out = out_f + NB * NACT + NB + NB;     // after policy, v1, v2
    attn_scatter_kernel<<<cdiv(NN * EMB, 256), 256, 0, s>>>(se, bsum2, BATCH, H,
                                                            pooled2, attn_out, NN);
    gemm(s, pooled2, EMB, 0, pj_w, EMB, 0, pj_b, gc2, EMB, 0, NB, EMB, EMB, 1);

    // ---- actor head -> softmax policy
    const float* cur = gc2;
    cur = run_block(s, cur, EMB, 2 * EMB, aB0, t0, t2, NB);
    cur = run_block(s, cur, 2 * EMB, EMB, aB1, t1, t2, NB);
    cur = run_block(s, cur, EMB, EMB, aB2, t0, t2, NB);
    cur = run_block(s, cur, EMB, EMB, aB3, t1, t2, NB);
    gemm(s, cur, EMB, 0, actor_fw, NACT, 0, actor_fb, logits, NACT, 0, NB, EMB, NACT, 1);
    softmax_rows_kernel<<<cdiv(NB, 8), wblk, 0, s>>>(logits, out_f, NB, NACT);

    // ---- critic heads -> values (N=1)
    cur = gc2;
    cur = run_block(s, cur, EMB, EMB, c1B0, t0, t2, NB);
    cur = run_block(s, cur, EMB, EMB, c1B1, t1, t2, NB);
    cur = run_block(s, cur, EMB, EMB, c1B2, t0, t2, NB);
    gemm(s, cur, EMB, 0, c1_fw, 1, 0, c1_fb, out_f + NB * NACT, 1, 0, NB, EMB, 1, 1);

    cur = gc2;
    cur = run_block(s, cur, EMB, EMB, c2B0, t0, t2, NB);
    cur = run_block(s, cur, EMB, EMB, c2B1, t1, t2, NB);
    cur = run_block(s, cur, EMB, EMB, c2B2, t0, t2, NB);
    gemm(s, cur, EMB, 0, c2_fw, 1, 0, c2_fb, out_f + NB * NACT + NB, 1, 0, NB, EMB, 1, 1);
}
